// MutualCrossAttention_81527069212975
// MI455X (gfx1250) — matrix operand order repaired and checked by reading, not compiled, not run on hardware
//
#include <hip/hip_runtime.h>
#include <hip/hip_bf16.h>

typedef __attribute__((ext_vector_type(16))) _Float16 v16h;
typedef __attribute__((ext_vector_type(8)))  _Float16 v8h;
typedef __attribute__((ext_vector_type(8)))  float    v8f;
typedef __attribute__((ext_vector_type(4)))  unsigned int v4ui;
typedef __attribute__((ext_vector_type(8)))  int v8i;
typedef __attribute__((ext_vector_type(4)))  int v4i;

constexpr int BATCH   = 4;
constexpr int HEADS_C = 8;
constexpr int DHEAD   = 64;
constexpr int N1C     = 2048;
constexpr int N2C     = 2048;
constexpr int NKV     = N1C + N2C;       // 4096
constexpr int DMODEL  = 1024;
constexpr int INNER   = HEADS_C * DHEAD; // 512

constexpr int PADK = 40;  // padded LDS stride for 32-wide K tiles (16B multiple)
constexpr int KSTR = 72;  // padded LDS stride for 64-wide tiles (16B multiple)

#if defined(__gfx1250__) && __has_builtin(__builtin_amdgcn_tensor_load_to_lds) && \
    __has_builtin(__builtin_amdgcn_s_wait_tensorcnt)
#define USE_TDM 1
#endif

__device__ __forceinline__ v16h make_frag(const _Float16* p0, const _Float16* p1) {
    union { v16h v; v8h h[2]; } u;
    u.h[0] = *(const v8h*)p0;
    u.h[1] = *(const v8h*)p1;
    return u.v;
}

// ---------------------------------------------------------------------------
// Projection GEMM: X[f32, 8192x1024] @ W[f32, 1024x512] -> f16 head-split
// out layout: [B, h, seq_out, 64], rows written at row_off (for k/v concat).
// Block: 256 thr (8 waves), tile 128x64, K-step 32, wave = 32x32 quad.
// oscale folds the attention 1/sqrt(d) into the Q projection.
// ---------------------------------------------------------------------------
__global__ __launch_bounds__(256) void proj_f32_to_heads(
    const float* __restrict__ X, const float* __restrict__ W,
    _Float16* __restrict__ out, int seq_out, int row_off, float oscale)
{
    __shared__ __align__(16) _Float16 Xl[128 * PADK];
    __shared__ __align__(16) _Float16 Wt[64 * PADK];   // stored transposed [n][k]

    const int tid  = threadIdx.x;
    const int wave = tid >> 5, lane = tid & 31;
    const int half = lane >> 4, ln = lane & 15;
    const int wm = wave & 3, wn = wave >> 2;           // 4 x 2 wave grid
    const int m0 = blockIdx.x * 128;
    const int n0 = blockIdx.y * 64;

    v8f acc[2][2] = {};

    for (int k0 = 0; k0 < DMODEL; k0 += 32) {
        #pragma unroll
        for (int i = 0; i < 16; ++i) {                 // 128x32 X tile, f32 -> f16
            int idx = tid + i * 256;
            int mm = idx >> 5, kk = idx & 31;
            Xl[mm * PADK + kk] = (_Float16)X[(size_t)(m0 + mm) * DMODEL + k0 + kk];
        }
        #pragma unroll
        for (int i = 0; i < 8; ++i) {                  // 32x64 W tile, transposed
            int idx = tid + i * 256;
            int kk = idx >> 6, nn = idx & 63;
            Wt[nn * PADK + kk] = (_Float16)W[(size_t)(k0 + kk) * INNER + n0 + nn];
        }
        __syncthreads();

        v16h a[2], b[2];
        #pragma unroll
        for (int ri = 0; ri < 2; ++ri) {
            const _Float16* base = Xl + (wm * 32 + ri * 16 + ln) * PADK;
            a[ri] = make_frag(base + half * 8, base + 16 + half * 8);
        }
        #pragma unroll
        for (int ci = 0; ci < 2; ++ci) {
            const _Float16* base = Wt + (wn * 32 + ci * 16 + ln) * PADK;
            b[ci] = make_frag(base + half * 8, base + 16 + half * 8);
        }
        #pragma unroll
        for (int ri = 0; ri < 2; ++ri)
            #pragma unroll
            for (int ci = 0; ci < 2; ++ci)
                acc[ri][ci] = __builtin_amdgcn_wmma_f32_16x16x32_f16(
                    false, a[ri], false, b[ci], (short)0, acc[ri][ci], false, false);
        __syncthreads();
    }

    #pragma unroll
    for (int ri = 0; ri < 2; ++ri)
        #pragma unroll
        for (int ci = 0; ci < 2; ++ci)
            #pragma unroll
            for (int r = 0; r < 8; ++r) {
                int m = m0 + wm * 32 + ri * 16 + half * 8 + r;
                int n = n0 + wn * 32 + ci * 16 + ln;
                int bq = m / N1C, srow = m % N1C;      // source seq len is 2048
                int head = n >> 6, dd = n & 63;
                out[(((size_t)(bq * HEADS_C + head)) * seq_out + row_off + srow) * DHEAD + dd]
                    = (_Float16)(acc[ri][ci][r] * oscale);
            }
}

// ---------------------------------------------------------------------------
// Flash attention. Block = (b,h) x 128 query rows. 8 waves x 16 rows.
// Key tile = 64. K tile fetched by the Tensor Data Mover straight into LDS
// (with per-row padding to the 72-halfword stride); V transposed by VALU.
// S = Q K^T, online softmax via shfl_xor over the 16-lane halves, P staged
// through per-wave LDS to become an A-fragment, O += P V.
// Q is pre-scaled by 1/sqrt(d) at projection time.
// ---------------------------------------------------------------------------
__global__ __launch_bounds__(256) void attn_flash(
    const _Float16* __restrict__ q, const _Float16* __restrict__ k,
    const _Float16* __restrict__ v, _Float16* __restrict__ o)
{
    __shared__ __align__(16) _Float16 Klds[64 * KSTR];     // [key][d], padded rows
    __shared__ __align__(16) _Float16 Vt[DHEAD * KSTR];    // transposed: [d][key]
    __shared__ __align__(16) _Float16 Plds[8 * 16 * DHEAD];

    const int tid  = threadIdx.x;
    const int wave = tid >> 5, lane = tid & 31;
    const int half = lane >> 4, ln = lane & 15;

    const int qblocks = N1C / 128;                         // 16
    const int bh = blockIdx.x / qblocks;
    const int q0 = (blockIdx.x % qblocks) * 128;
    const int bq = bh / HEADS_C, head = bh % HEADS_C;

    const _Float16* qbase = q + (size_t)bh * N1C * DHEAD;
    const _Float16* kbase = k + (size_t)bh * NKV * DHEAD;
    const _Float16* vbase = v + (size_t)bh * NKV * DHEAD;

    // Q fragments for this wave's 16 rows (d split into two 32-wide chunks)
    const int qrow = q0 + wave * 16 + ln;
    v16h qa[2];
    #pragma unroll
    for (int c = 0; c < 2; ++c) {
        const _Float16* base = qbase + (size_t)qrow * DHEAD + c * 32;
        qa[c] = make_frag(base + half * 8, base + 16 + half * 8);
    }

    float m_run[8], l_run[8];
    v8f oacc[4] = {};
    #pragma unroll
    for (int r = 0; r < 8; ++r) { m_run[r] = -1e30f; l_run[r] = 0.f; }

    _Float16* Pw = Plds + wave * 16 * DHEAD;

    for (int j0 = 0; j0 < NKV; j0 += 64) {
#ifdef USE_TDM
        if (wave == 0) {
            // D# for a 2D tile: 64 rows x 64 f16, row stride 64 elements in
            // memory, LDS padded by 4 DWORDs every 32 DWORDs (-> 72 halfwords).
            unsigned long long ga =
                (unsigned long long)(uintptr_t)(kbase + (size_t)j0 * DHEAD);
            unsigned int lds_off = (unsigned int)(uintptr_t)&Klds[0];
            v4ui g0 = {};
            g0[0] = 1u;                                   // count=1, user mode
            g0[1] = lds_off;                              // lds_addr
            g0[2] = (unsigned int)ga;                     // global_addr[31:0]
            g0[3] = ((unsigned int)(ga >> 32) & 0x1FFFFFFu) | (2u << 30); // [56:32] | type=2
            v8i g1 = {};
            g1[0] = (1 << 16)      // data_size = 2 bytes
                  | (1 << 20)      // pad_enable
                  | (4 << 22)      // pad_interval: 32 DWORDs
                  | (3 << 25);     // pad_amount: 4 DWORDs
            g1[1] = (int)((64u & 0xFFFFu) << 16);         // tensor_dim0 = 64 (lo16)
            g1[2] = (int)((64u >> 16) | ((unsigned)NKV & 0xFFFFu) << 16); // dim0 hi | dim1 lo
            g1[3] = (int)(((unsigned)NKV >> 16) | (64u << 16));           // dim1 hi | tile_dim0
            g1[4] = 64;                                   // tile_dim1 = 64, tile_dim2 = 0
            g1[5] = 64;                                   // tensor_dim0_stride = 64
            g1[6] = 0;
            g1[7] = 0;
            v4i g2 = {};
            v4i g3 = {};
#if __clang_major__ >= 23
            v8i g4 = {};
            __builtin_amdgcn_tensor_load_to_lds(g0, g1, g2, g3, g4, 0);
#else
            __builtin_amdgcn_tensor_load_to_lds(g0, g1, g2, g3, 0);
#endif
        }
        #pragma unroll
        for (int i = 0; i < 16; ++i) {                    // V tile, transposed
            int idx = tid + i * 256;
            int kr = idx >> 6, kc = idx & 63;
            Vt[kc * KSTR + kr] = vbase[(size_t)(j0 + kr) * DHEAD + kc];
        }
        if (j0 + 64 < NKV && tid < 64) {                  // prefetch next tiles
            __builtin_prefetch(kbase + (size_t)(j0 + 64 + tid) * DHEAD, 0, 1);
            __builtin_prefetch(vbase + (size_t)(j0 + 64 + tid) * DHEAD, 0, 1);
        }
        if (wave == 0) __builtin_amdgcn_s_wait_tensorcnt(0);
        __syncthreads();
#else
        #pragma unroll
        for (int i = 0; i < 16; ++i) {                    // 64x64 K and V tiles
            int idx = tid + i * 256;
            int kr = idx >> 6, kc = idx & 63;
            Klds[kr * KSTR + kc] = kbase[(size_t)(j0 + kr) * DHEAD + kc];
            Vt[kc * KSTR + kr]   = vbase[(size_t)(j0 + kr) * DHEAD + kc];
        }
        __syncthreads();
#endif

        // S = Q K^T for four 16-key subtiles (Q already carries 1/sqrt(d))
        v8f s[4];
        #pragma unroll
        for (int t = 0; t < 4; ++t) {
            const _Float16* kp = Klds + (t * 16 + ln) * KSTR;
            v16h b0 = make_frag(kp + half * 8,      kp + 16 + half * 8);
            v16h b1 = make_frag(kp + 32 + half * 8, kp + 48 + half * 8);
            v8f a = {};
            a = __builtin_amdgcn_wmma_f32_16x16x32_f16(false, qa[0], false, b0, (short)0, a, false, false);
            a = __builtin_amdgcn_wmma_f32_16x16x32_f16(false, qa[1], false, b1, (short)0, a, false, false);
            s[t] = a;
        }

        // online softmax (rows = half*8 + r, cols spread over 16 lanes/half)
        #pragma unroll
        for (int r = 0; r < 8; ++r) {
            float s0 = s[0][r], s1 = s[1][r], s2 = s[2][r], s3 = s[3][r];
            float mx = fmaxf(fmaxf(s0, s1), fmaxf(s2, s3));
            #pragma unroll
            for (int msk = 1; msk < 16; msk <<= 1)
                mx = fmaxf(mx, __shfl_xor(mx, msk, 32));
            float mnew = fmaxf(m_run[r], mx);
            float corr = __expf(m_run[r] - mnew);
            float p0 = __expf(s0 - mnew), p1 = __expf(s1 - mnew);
            float p2 = __expf(s2 - mnew), p3 = __expf(s3 - mnew);
            float rsum = p0 + p1 + p2 + p3;
            #pragma unroll
            for (int msk = 1; msk < 16; msk <<= 1)
                rsum += __shfl_xor(rsum, msk, 32);
            l_run[r] = l_run[r] * corr + rsum;
            m_run[r] = mnew;
            #pragma unroll
            for (int t = 0; t < 4; ++t) oacc[t][r] *= corr;
            // stash P into this wave's LDS patch (C-layout -> memory)
            _Float16* pr = Pw + (half * 8 + r) * DHEAD + ln;
            pr[0]  = (_Float16)p0;
            pr[16] = (_Float16)p1;
            pr[32] = (_Float16)p2;
            pr[48] = (_Float16)p3;
        }

        // A-fragments of P (contraction over the 64 keys, two 32-chunks)
        v16h pa[2];
        #pragma unroll
        for (int c = 0; c < 2; ++c) {
            const _Float16* base = Pw + ln * DHEAD + c * 32;
            pa[c] = make_frag(base + half * 8, base + 16 + half * 8);
        }

        // O += P V  (four 16-wide d subtiles)
        #pragma unroll
        for (int dt = 0; dt < 4; ++dt) {
            const _Float16* vp = Vt + (dt * 16 + ln) * KSTR;
            v16h vb0 = make_frag(vp + half * 8,      vp + 16 + half * 8);
            v16h vb1 = make_frag(vp + 32 + half * 8, vp + 48 + half * 8);
            v8f a = oacc[dt];
            a = __builtin_amdgcn_wmma_f32_16x16x32_f16(false, pa[0], false, vb0, (short)0, a, false, false);
            a = __builtin_amdgcn_wmma_f32_16x16x32_f16(false, pa[1], false, vb1, (short)0, a, false, false);
            oacc[dt] = a;
        }
        __syncthreads();
    }

    // normalize and store to [B, N1, h*d] (f16, feeds the output projection)
    #pragma unroll
    for (int dt = 0; dt < 4; ++dt)
        #pragma unroll
        for (int r = 0; r < 8; ++r) {
            float val = oacc[dt][r] / l_run[r];
            int row = q0 + wave * 16 + half * 8 + r;
            int col = head * DHEAD + dt * 16 + ln;
            o[((size_t)(bq * N1C + row)) * INNER + col] = (_Float16)val;
        }
}

// ---------------------------------------------------------------------------
// Output projection: A[f16, 8192x512] @ Wo[f32, 512x1024] + bo -> f32 out
// ---------------------------------------------------------------------------
__global__ __launch_bounds__(256) void out_proj(
    const _Float16* __restrict__ A, const float* __restrict__ W,
    const float* __restrict__ bias, float* __restrict__ out)
{
    __shared__ __align__(16) _Float16 Al[128 * PADK];
    __shared__ __align__(16) _Float16 Wt[64 * PADK];       // transposed [n][k]

    const int tid  = threadIdx.x;
    const int wave = tid >> 5, lane = tid & 31;
    const int half = lane >> 4, ln = lane & 15;
    const int wm = wave & 3, wn = wave >> 2;
    const int m0 = blockIdx.x * 128;
    const int n0 = blockIdx.y * 64;

    v8f acc[2][2] = {};

    for (int k0 = 0; k0 < INNER; k0 += 32) {
        #pragma unroll
        for (int i = 0; i < 16; ++i) {
            int idx = tid + i * 256;
            int mm = idx >> 5, kk = idx & 31;
            Al[mm * PADK + kk] = A[(size_t)(m0 + mm) * INNER + k0 + kk];
        }
        #pragma unroll
        for (int i = 0; i < 8; ++i) {
            int idx = tid + i * 256;
            int kk = idx >> 6, nn = idx & 63;
            Wt[nn * PADK + kk] = (_Float16)W[(size_t)(k0 + kk) * DMODEL + n0 + nn];
        }
        __syncthreads();

        v16h a[2], b[2];
        #pragma unroll
        for (int ri = 0; ri < 2; ++ri) {
            const _Float16* base = Al + (wm * 32 + ri * 16 + ln) * PADK;
            a[ri] = make_frag(base + half * 8, base + 16 + half * 8);
        }
        #pragma unroll
        for (int ci = 0; ci < 2; ++ci) {
            const _Float16* base = Wt + (wn * 32 + ci * 16 + ln) * PADK;
            b[ci] = make_frag(base + half * 8, base + 16 + half * 8);
        }
        #pragma unroll
        for (int ri = 0; ri < 2; ++ri)
            #pragma unroll
            for (int ci = 0; ci < 2; ++ci)
                acc[ri][ci] = __builtin_amdgcn_wmma_f32_16x16x32_f16(
                    false, a[ri], false, b[ci], (short)0, acc[ri][ci], false, false);
        __syncthreads();
    }

    #pragma unroll
    for (int ri = 0; ri < 2; ++ri)
        #pragma unroll
        for (int ci = 0; ci < 2; ++ci)
            #pragma unroll
            for (int r = 0; r < 8; ++r) {
                int m = m0 + wm * 32 + ri * 16 + half * 8 + r;
                int n = n0 + wn * 32 + ci * 16 + ln;
                out[(size_t)m * DMODEL + n] = acc[ri][ci][r] + bias[n];
            }
}

// ---------------------------------------------------------------------------
extern "C" void kernel_launch(void* const* d_in, const int* in_sizes, int n_in,
                              void* d_out, int out_size, void* d_ws, size_t ws_size,
                              hipStream_t stream) {
    const float* x1  = (const float*)d_in[0];
    const float* x2  = (const float*)d_in[1];
    const float* Wq1 = (const float*)d_in[2];
    const float* Wk1 = (const float*)d_in[3];
    const float* Wv1 = (const float*)d_in[4];
    const float* Wk2 = (const float*)d_in[5];
    const float* Wv2 = (const float*)d_in[6];
    const float* Wo  = (const float*)d_in[7];
    const float* bo  = (const float*)d_in[8];
    float* out = (float*)d_out;

    // workspace: q(8MB) + k(16MB) + v(16MB) + attn(8MB) f16
    _Float16* qf   = (_Float16*)d_ws;
    _Float16* kf   = qf + (size_t)BATCH * HEADS_C * N1C * DHEAD;
    _Float16* vf   = kf + (size_t)BATCH * HEADS_C * NKV * DHEAD;
    _Float16* attn = vf + (size_t)BATCH * HEADS_C * NKV * DHEAD;

    dim3 blk(256);
    dim3 gproj((BATCH * N1C) / 128, INNER / 64);           // 64 x 8

    proj_f32_to_heads<<<gproj, blk, 0, stream>>>(x1, Wq1, qf, N1C, 0, 0.125f);
    proj_f32_to_heads<<<gproj, blk, 0, stream>>>(x1, Wk1, kf, NKV, 0, 1.0f);
    proj_f32_to_heads<<<gproj, blk, 0, stream>>>(x1, Wv1, vf, NKV, 0, 1.0f);
    proj_f32_to_heads<<<gproj, blk, 0, stream>>>(x2, Wk2, kf, NKV, N1C, 1.0f);
    proj_f32_to_heads<<<gproj, blk, 0, stream>>>(x2, Wv2, vf, NKV, N1C, 1.0f);

    dim3 gattn(BATCH * HEADS_C * (N1C / 128));             // 512
    attn_flash<<<gattn, blk, 0, stream>>>(qf, kf, vf, attn);

    dim3 goproj((BATCH * N1C) / 128, DMODEL / 64);         // 64 x 16
    out_proj<<<goproj, blk, 0, stream>>>(attn, Wo, bo, out);
}
